// DLGNN_53678501266027
// MI455X (gfx1250) — matrix operand order repaired and checked by reading, not compiled, not run on hardware
//
#include <hip/hip_runtime.h>
#include <stdint.h>
#include <stddef.h>

typedef __attribute__((ext_vector_type(16))) _Float16 v16h;
typedef __attribute__((ext_vector_type(8)))  float    v8f;

union FragH { v16h v; unsigned int u[8]; };
union HBits { _Float16 h; unsigned short s; };

#define CDIV(a,b) (((a)+(b)-1)/(b))

#ifndef USE_ASYNC_LDS
#define USE_ASYNC_LDS 1
#endif

#define BN   2
#define ECH  64
#define NFC  256
#define NQI  47
#define NQ   2209
#define NQP  2224      // 139*16
#define NKI  46
#define NK   2116
#define NKP  2128      // 133*16
#define DDIM 576

// ---------------- elementwise / prep kernels ----------------

__global__ void k_preproc(const float* __restrict__ src, float* __restrict__ dst,
                          int n, int hw, int mode) {
  int i = blockIdx.x * blockDim.x + threadIdx.x;
  if (i >= n) return;
  int c = (i / hw) % 3;
  const float vm[3] = {0.485f, 0.456f, 0.406f};
  const float vs[3] = {0.229f, 0.224f, 0.225f};
  const float rm[3] = {0.4488f, 0.4371f, 0.4040f};
  float v = src[i];
  dst[i] = (mode == 0) ? ((v * (1.0f/255.0f) - vm[c]) / vs[c]) : (v - 255.0f * rm[c]);
}

__global__ void k_conv3_direct(const float* __restrict__ in, const float* __restrict__ w,
                               const float* __restrict__ bias, float* __restrict__ out,
                               int Co, int H, int W, int relu) {
  int i = blockIdx.x * blockDim.x + threadIdx.x;
  int tot = BN * Co * H * W;
  if (i >= tot) return;
  int x  = i % W;
  int y  = (i / W) % H;
  int co = (i / (W * H)) % Co;
  int b  = i / (W * H * Co);
  float acc = bias[co];
  for (int ci = 0; ci < 3; ++ci)
    for (int dy = 0; dy < 3; ++dy)
      for (int dx = 0; dx < 3; ++dx) {
        int yy = y + dy - 1, xx = x + dx - 1;
        if ((unsigned)yy < (unsigned)H && (unsigned)xx < (unsigned)W)
          acc += w[((co*3 + ci)*3 + dy)*3 + dx] *
                 in[(((size_t)b*3 + ci)*H + yy)*(size_t)W + xx];
      }
  if (relu) acc = fmaxf(acc, 0.0f);
  out[i] = acc;
}

__global__ void k_f32_to_f16(const float* __restrict__ src, unsigned short* __restrict__ dst,
                             int n, int relu) {
  int i = blockIdx.x * blockDim.x + threadIdx.x;
  if (i >= n) return;
  float v = src[i];
  if (relu) v = fmaxf(v, 0.0f);
  HBits cv; cv.h = (_Float16)v;
  dst[i] = cv.s;
}

// pack conv weights [Co,Ci,3,3] f32 -> f16.
// K-order: k = (cc*9 + t)*32 + l  with ci = cc*32+l, tap t (ci-chunk outer, tap inner).
// pair-interleaved: dst[((k>>1)*Cout + co)*2 + (k&1)]
__global__ void k_pack(const float* __restrict__ w, unsigned short* __restrict__ dst,
                       int Cin, int Cout) {
  int i = blockIdx.x * blockDim.x + threadIdx.x;
  int tot = Cin * 9 * Cout;
  if (i >= tot) return;
  int co = i % Cout;
  int k  = i / Cout;
  int l  = k & 31;
  int t  = (k >> 5) % 9;
  int cc = k / 288;
  int ci = cc * 32 + l;
  int dy = t / 3, dx = t % 3;
  float v = w[(((size_t)co*Cin + ci)*3 + dy)*3 + dx];
  HBits cv; cv.h = (_Float16)v;
  dst[((size_t)(k >> 1) * Cout + co) * 2 + (k & 1)] = cv.s;
}

// ---------------- WMMA implicit-GEMM 3x3 SAME conv ----------------
// block = 128 thr (4 waves), one row-segment of 16 pixels, 4*COPW out channels.
// Per 32-ci chunk: stage 3 rows x 32 ci x 20 cols (f16) into LDS (async-to-LDS when
// available), then 9 taps of WMMA read A-fragments from LDS.  Each wave: 16px x COPW co.
template<int CIN, int COPW>
__global__ void __launch_bounds__(128)
k_conv_wmma(const unsigned short* __restrict__ inH, const unsigned short* __restrict__ wpk,
            const float* __restrict__ bias, const float* __restrict__ res, float scale,
            float* __restrict__ out, int Cout, int H, int W, int relu, int coGroups) {
  __shared__ __align__(16) unsigned short At2[3 * 32 * 20];   // 3840 B
  int b  = blockIdx.z;
  int y  = blockIdx.y;
  int cg = blockIdx.x % coGroups;
  int x0 = (blockIdx.x / coGroups) * 16;
  int tid  = threadIdx.x;
  int lane = tid & 31;
  int wv   = tid >> 5;
  int half = lane >> 4;
  int lm   = lane & 15;
  int co0  = (cg * 4 + wv) * COPW;
  v8f acc0 = {};
  v8f acc1 = {};
  for (int cc = 0; cc < CIN / 32; ++cc) {
    __syncthreads();                       // previous-chunk LDS reads done
    for (int e = tid; e < 960; e += 128) { // 960 dwords = 3*32*10
      int col2 = e % 10;
      int ci32 = (e / 10) % 32;
      int r    = e / 320;
      int yy = y + r - 1;
      int xg = x0 - 2 + col2 * 2;          // even; dword fully in or out of range
      unsigned short* lp = &At2[(r * 32 + ci32) * 20 + col2 * 2];
      if ((unsigned)yy < (unsigned)H && (unsigned)xg < (unsigned)W) {
        const unsigned short* gp =
            &inH[(((size_t)b * CIN + (cc * 32 + ci32)) * H + yy) * (size_t)W + xg];
#if USE_ASYNC_LDS
        unsigned lofs = (unsigned)(size_t)lp;
        unsigned long long ga = (unsigned long long)(size_t)gp;
        asm volatile("global_load_async_to_lds_b32 %0, %1, off"
                     :: "v"(lofs), "v"(ga) : "memory");
#else
        *(unsigned int*)lp = *(const unsigned int*)gp;
#endif
      } else {
        *(unsigned int*)lp = 0u;
      }
    }
#if USE_ASYNC_LDS
    asm volatile("s_wait_asynccnt 0" ::: "memory");
#endif
    __syncthreads();                       // staged tile visible to all waves
    if (cc + 1 < CIN / 32)
      __builtin_prefetch((const void*)&wpk[(size_t)((cc + 1) * 144) * Cout * 2 +
                                           (size_t)co0 * 2], 0, 1);
#pragma unroll
    for (int t = 0; t < 9; ++t) {
      int rA   = t / 3;
      int colA = lm + (t % 3) + 1;         // m + dx + 2
      FragH a, b0, b1;
#pragma unroll
      for (int v = 0; v < 8; ++v) {
        int k0 = ((v & 4) ? 16 : 0) + half * 8 + (v & 3) * 2;
        unsigned lo = At2[(rA * 32 + k0)     * 20 + colA];
        unsigned hi = At2[(rA * 32 + k0 + 1) * 20 + colA];
        a.u[v] = lo | (hi << 16);
      }
      size_t wrow = ((size_t)((cc * 9 + t) * 16)) * Cout * 2;  // (kb>>1)*Cout*2
#pragma unroll
      for (int v = 0; v < 8; ++v) {
        int kr = ((v & 4) ? 16 : 0) + half * 8 + (v & 3) * 2;
        size_t o = wrow + (size_t)(kr >> 1) * Cout * 2;
        b0.u[v] = *(const unsigned int*)&wpk[o + (size_t)(co0 + lm) * 2];
        if (COPW == 32)
          b1.u[v] = *(const unsigned int*)&wpk[o + (size_t)(co0 + 16 + lm) * 2];
      }
      acc0 = __builtin_amdgcn_wmma_f32_16x16x32_f16(false, a.v, false, b0.v,
                                                    (short)0, acc0, false, false);
      if (COPW == 32)
        acc1 = __builtin_amdgcn_wmma_f32_16x16x32_f16(false, a.v, false, b1.v,
                                                      (short)0, acc1, false, false);
    }
  }
  {
    int co = co0 + lm;
    float bsv = bias[co];
#pragma unroll
    for (int r = 0; r < 8; ++r) {
      int mm = r + 8 * half;
      size_t oidx = (((size_t)b * Cout + co) * H + y) * (size_t)W + x0 + mm;
      float vv = acc0[r] + bsv;
      if (res)  vv = res[oidx] + scale * vv;
      if (relu) vv = fmaxf(vv, 0.0f);
      out[oidx] = vv;
    }
  }
  if (COPW == 32) {
    int co = co0 + 16 + lm;
    float bsv = bias[co];
#pragma unroll
    for (int r = 0; r < 8; ++r) {
      int mm = r + 8 * half;
      size_t oidx = (((size_t)b * Cout + co) * H + y) * (size_t)W + x0 + mm;
      float vv = acc1[r] + bsv;
      if (res)  vv = res[oidx] + scale * vv;
      if (relu) vv = fmaxf(vv, 0.0f);
      out[oidx] = vv;
    }
  }
}

// ---------------- patch banks, QK GEMM, top-k/gate ----------------

__global__ void k_extract_q(const float* __restrict__ E, unsigned short* __restrict__ Qb,
                            float* __restrict__ qn2) {
  int i = blockIdx.x * blockDim.x + threadIdx.x;
  if (i >= BN * NQP) return;
  int q = i % NQP, b = i / NQP;
  unsigned short* dst = Qb + (size_t)i * DDIM;
  if (q >= NQ) {
    for (int d = 0; d < DDIM; ++d) dst[d] = 0;
    qn2[i] = 0.0f;
    return;
  }
  int qi = q / NQI, qj = q % NQI;
  float s = 0.0f;
  for (int c = 0; c < ECH; ++c)
    for (int di = 0; di < 3; ++di)
      for (int dj = 0; dj < 3; ++dj) {
        float v = E[(((size_t)b * ECH + c) * 96 + (2*qi + di)) * 96 + (2*qj + dj)];
        s += v * v;
        HBits cv; cv.h = (_Float16)v;
        dst[c * 9 + di * 3 + dj] = cv.s;
      }
  qn2[i] = s;
}

__global__ void k_extract_k(const float* __restrict__ E, unsigned short* __restrict__ Kb,
                            float* __restrict__ kn2) {
  int i = blockIdx.x * blockDim.x + threadIdx.x;
  if (i >= BN * NKP) return;
  int n = i % NKP, b = i / NKP;
  unsigned short* dst = Kb + (size_t)i * DDIM;
  if (n >= NK) {
    for (int d = 0; d < DDIM; ++d) dst[d] = 0;
    kn2[i] = 0.0f;
    return;
  }
  int ki = n / NKI, kj = n % NKI;
  float s = 0.0f;
  for (int c = 0; c < ECH; ++c)
    for (int di = 0; di < 3; ++di)
      for (int dj = 0; dj < 3; ++dj) {
        float v = E[(((size_t)b * ECH + c) * 48 + (ki + di)) * 48 + (kj + dj)];
        s += v * v;
        HBits cv; cv.h = (_Float16)v;
        dst[c * 9 + di * 3 + dj] = cv.s;
      }
  kn2[i] = s;
}

// one wave per 16x16 tile of S = Q . K^T (f32 accum); grid (NKP/16, NQP/16, BN)
__global__ void __launch_bounds__(32)
k_qk_wmma(const unsigned short* __restrict__ Qb, const unsigned short* __restrict__ Kb,
          float* __restrict__ S) {
  int nt = blockIdx.x;
  int qt = blockIdx.y;
  int b  = blockIdx.z;
  int lane = threadIdx.x;
  int half = lane >> 4;
  int lm   = lane & 15;
  const unsigned short* qrow = Qb + ((size_t)b * NQP + qt * 16 + lm) * DDIM;
  const unsigned short* krow = Kb + ((size_t)b * NKP + nt * 16 + lm) * DDIM;
  v8f acc = {};
#pragma unroll
  for (int s = 0; s < DDIM / 32; ++s) {
    int kb = s << 5;
    FragH a, bb;
#pragma unroll
    for (int v = 0; v < 8; ++v) {
      int k0 = ((v & 4) ? 16 : 0) + half * 8 + (v & 3) * 2;
      a.u[v] = *(const unsigned int*)&qrow[kb + k0];
    }
#pragma unroll
    for (int v = 0; v < 8; ++v) {
      int kr = ((v & 4) ? 16 : 0) + half * 8 + (v & 3) * 2;
      bb.u[v] = *(const unsigned int*)&krow[kb + kr];
    }
    acc = __builtin_amdgcn_wmma_f32_16x16x32_f16(false, a.v, false, bb.v,
                                                 (short)0, acc, false, false);
  }
#pragma unroll
  for (int r = 0; r < 8; ++r)
    S[((size_t)b * NQP + qt * 16 + r + 8 * half) * NKP + nt * 16 + lm] = acc[r];
}

__global__ void k_topk(const float* __restrict__ S, const float* __restrict__ qn2,
                       const float* __restrict__ kn2,
                       const unsigned short* __restrict__ Qb, const unsigned short* __restrict__ Kb,
                       const float* __restrict__ gw, const float* __restrict__ gb,
                       int* __restrict__ idx, float* __restrict__ wk) {
  int i = blockIdx.x * blockDim.x + threadIdx.x;
  if (i >= BN * NQ) return;
  int q = i % NQ, b = i / NQ;
  const float* srow = S + ((size_t)b * NQP + q) * NKP;
  float qn = qn2[b * NQP + q];
  float sc[5]; int id[5];
  for (int k = 0; k < 5; ++k) { sc[k] = -3.4e38f; id[k] = 0; }
  for (int n = 0; n < NK; ++n) {
    float s = -(qn + kn2[b * NKP + n] - 2.0f * srow[n]);
    if (s > sc[4]) {
      int p = 4;
      while (p > 0 && s > sc[p - 1]) { sc[p] = sc[p - 1]; id[p] = id[p - 1]; --p; }
      sc[p] = s; id[p] = n;
    }
  }
  float e[5], wsum = 0.0f;
  for (int k = 0; k < 5; ++k) { e[k] = expf((sc[k] - sc[0]) * (1.0f / DDIM)); wsum += e[k]; }
  const unsigned short* qr = Qb + ((size_t)b * NQP + q) * DDIM;
  for (int k = 0; k < 5; ++k) {
    int n = id[k];
    const unsigned short* kr = Kb + ((size_t)b * NKP + n) * DDIM;
    float z = 0.0f;
    for (int d = 0; d < DDIM; ++d) {
      HBits a, c; a.s = qr[d]; c.s = kr[d];
      z += ((float)a.h - (float)c.h) * gw[d];
    }
    z += gb[0];
    float g = 1.0f / (1.0f + expf(-z));
    wk[(size_t)i * 5 + k]  = (e[k] / wsum) * g;
    idx[(size_t)i * 5 + k] = n;
  }
}

// ---------------- GCN aggregation (gather form of fold) ----------------

__global__ void k_downsample(const float* __restrict__ in, float* __restrict__ out) {
  int i = blockIdx.x * blockDim.x + threadIdx.x;
  int tot = BN * NFC * 48 * 48;
  if (i >= tot) return;
  int x = i % 48, y = (i / 48) % 48;
  int c = (i / (48 * 48)) % NFC, b = i / (48 * 48 * NFC);
  const float* p = in + (((size_t)b * NFC + c) * 96 + 2 * y) * 96 + 2 * x;
  out[i] = 0.25f * (p[0] + p[1] + p[96] + p[97]);
}

__global__ void k_gcn_lr(const float* __restrict__ x1ds, const int* __restrict__ idx,
                         const float* __restrict__ wk, float* __restrict__ out) {
  int i = blockIdx.x * blockDim.x + threadIdx.x;
  int tot = BN * NFC * 96 * 96;
  if (i >= tot) return;
  int x = i % 96, y = (i / 96) % 96;
  int c = (i / (96 * 96)) % NFC, b = i / (96 * 96 * NFC);
  float acc = 0.0f, cnt = 0.0f;
  for (int qi = y / 2 - 1; qi <= y / 2; ++qi) {
    if ((unsigned)qi >= (unsigned)NQI) continue;
    int di = y - 2 * qi; if ((unsigned)di >= 3u) continue;
    for (int qj = x / 2 - 1; qj <= x / 2; ++qj) {
      if ((unsigned)qj >= (unsigned)NQI) continue;
      int dj = x - 2 * qj; if ((unsigned)dj >= 3u) continue;
      int q = qi * NQI + qj;
      cnt += 1.0f;
      const int*   ip = idx + ((size_t)b * NQ + q) * 5;
      const float* wp = wk  + ((size_t)b * NQ + q) * 5;
      for (int k = 0; k < 5; ++k) {
        int n = ip[k]; int ki = n / NKI, kj = n % NKI;
        acc += wp[k] * x1ds[(((size_t)b * NFC + c) * 48 + (ki + di)) * 48 + (kj + dj)];
      }
    }
  }
  out[i] = acc / fmaxf(cnt, 1.0f);
}

__global__ void k_gcn_hr(const float* __restrict__ x1, const int* __restrict__ idx,
                         const float* __restrict__ wk, float* __restrict__ out) {
  int i = blockIdx.x * blockDim.x + threadIdx.x;
  int tot = BN * NFC * 192 * 192;
  if (i >= tot) return;
  int x = i % 192, y = (i / 192) % 192;
  int c = (i / (192 * 192)) % NFC, b = i / (192 * 192 * NFC);
  float acc = 0.0f, cnt = 0.0f;
  for (int qi = y / 4 - 1; qi <= y / 4; ++qi) {
    if ((unsigned)qi >= (unsigned)NQI) continue;
    int di = y - 4 * qi; if ((unsigned)di >= 6u) continue;
    for (int qj = x / 4 - 1; qj <= x / 4; ++qj) {
      if ((unsigned)qj >= (unsigned)NQI) continue;
      int dj = x - 4 * qj; if ((unsigned)dj >= 6u) continue;
      int q = qi * NQI + qj;
      cnt += 1.0f;
      const int*   ip = idx + ((size_t)b * NQ + q) * 5;
      const float* wp = wk  + ((size_t)b * NQ + q) * 5;
      for (int k = 0; k < 5; ++k) {
        int n = ip[k]; int ki = n / NKI, kj = n % NKI;
        acc += wp[k] * x1[(((size_t)b * NFC + c) * 96 + (2 * ki + di)) * 96 + (2 * kj + dj)];
      }
    }
  }
  out[i] = acc / fmaxf(cnt, 1.0f);
}

// ---------------- misc tail-end kernels ----------------

__global__ void k_concat_f16(const float* __restrict__ a, const float* __restrict__ b2,
                             unsigned short* __restrict__ dst, int H, int W) {
  int i = blockIdx.x * blockDim.x + threadIdx.x;
  int hw = H * W;
  int tot = BN * 512 * hw;
  if (i >= tot) return;
  int p = i % hw;
  int c = (i / hw) % 512;
  int b = i / (hw * 512);
  float v = (c < 256) ? a[((size_t)b * 256 + c) * hw + p]
                      : b2[((size_t)b * 256 + (c - 256)) * hw + p];
  HBits cv; cv.h = (_Float16)v;
  dst[i] = cv.s;
}

__global__ void k_pixel_shuffle(const float* __restrict__ in, float* __restrict__ out) {
  int i = blockIdx.x * blockDim.x + threadIdx.x;
  int tot = BN * NFC * 192 * 192;
  if (i >= tot) return;
  int x = i % 192, y = (i / 192) % 192;
  int c = (i / (192 * 192)) % NFC, b = i / (192 * 192 * NFC);
  int ci = c * 4 + (y & 1) * 2 + (x & 1);
  out[i] = in[(((size_t)b * 1024 + ci) * 96 + (y >> 1)) * 96 + (x >> 1)];
}

__global__ void k_tail(const float* __restrict__ in, const float* __restrict__ w,
                       const float* __restrict__ bias, float* __restrict__ out) {
  int i = blockIdx.x * blockDim.x + threadIdx.x;
  int tot = BN * 3 * 192 * 192;
  if (i >= tot) return;
  int x = i % 192, y = (i / 192) % 192;
  int co = (i / (192 * 192)) % 3, b = i / (192 * 192 * 3);
  const float rm[3] = {0.4488f, 0.4371f, 0.4040f};
  float acc = bias[co];
  for (int ci = 0; ci < 256; ++ci)
    for (int dy = 0; dy < 3; ++dy)
      for (int dx = 0; dx < 3; ++dx) {
        int yy = y + dy - 1, xx = x + dx - 1;
        if ((unsigned)yy < 192u && (unsigned)xx < 192u)
          acc += w[(((size_t)co * 256 + ci) * 3 + dy) * 3 + dx] *
                 in[(((size_t)b * 256 + ci) * 192 + yy) * 192 + xx];
      }
  out[i] = acc + 255.0f * rm[co];
}

// ---------------- host orchestration ----------------

extern "C" void kernel_launch(void* const* d_in, const int* in_sizes, int n_in,
                              void* d_out, int out_size, void* d_ws, size_t ws_size,
                              hipStream_t stream) {
  (void)in_sizes; (void)n_in; (void)out_size; (void)ws_size;
  const float* x     = (const float*)d_in[0];
  const float* x_son = (const float*)d_in[1];
  const float* ew1 = (const float*)d_in[2];
  const float* eb1 = (const float*)d_in[3];
  const float* ew2 = (const float*)d_in[4];
  const float* eb2 = (const float*)d_in[5];
  const float* gw  = (const float*)d_in[6];
  const float* gb  = (const float*)d_in[7];
  const float* hdw = (const float*)d_in[8];
  const float* hdb = (const float*)d_in[9];
  const float* b1w = (const float*)d_in[10];
  const float* b1b = (const float*)d_in[11];
  const float* fbw1 = (const float*)d_in[12];
  const float* fbb1 = (const float*)d_in[13];
  const float* fbw2 = (const float*)d_in[14];
  const float* fbb2 = (const float*)d_in[15];
  const float* b2w = (const float*)d_in[16];
  const float* b2b = (const float*)d_in[17];
  const float* b2tw = (const float*)d_in[18];
  const float* b2tb = (const float*)d_in[19];
  const float* upw = (const float*)d_in[20];
  const float* upb = (const float*)d_in[21];
  const float* fuw1 = (const float*)d_in[22];
  const float* fub1 = (const float*)d_in[23];
  const float* fuw2 = (const float*)d_in[24];
  const float* fub2 = (const float*)d_in[25];
  const float* tw  = (const float*)d_in[26];
  const float* tb  = (const float*)d_in[27];
  float* outp = (float*)d_out;

  char* ws = (char*)d_ws;
  size_t off = 0;
  auto alloc = [&](size_t bytes) -> char* {
    char* p = ws + off;
    off += (bytes + 255) & ~(size_t)255;
    return p;
  };

  const size_t HW96 = 96 * 96, HW48 = 48 * 48, HW192 = 192 * 192;

  // packed weights
  unsigned short* pk_e2 = (unsigned short*)alloc((size_t)ECH * 9 * ECH * 2);
  unsigned short* pk_b1[16];
  for (int i = 0; i < 16; ++i) pk_b1[i] = (unsigned short*)alloc((size_t)NFC * 9 * NFC * 2);
  unsigned short* pk_fb1 = (unsigned short*)alloc((size_t)512 * 9 * NFC * 2);
  unsigned short* pk_fb2 = (unsigned short*)alloc((size_t)NFC * 9 * NFC * 2);
  unsigned short* pk_b2[16];
  for (int i = 0; i < 16; ++i) pk_b2[i] = (unsigned short*)alloc((size_t)NFC * 9 * NFC * 2);
  unsigned short* pk_b2t = (unsigned short*)alloc((size_t)NFC * 9 * NFC * 2);
  unsigned short* pk_up  = (unsigned short*)alloc((size_t)NFC * 9 * 1024 * 2);
  unsigned short* pk_fu1 = (unsigned short*)alloc((size_t)512 * 9 * NFC * 2);
  unsigned short* pk_fu2 = (unsigned short*)alloc((size_t)NFC * 9 * NFC * 2);
  // embed / graph buffers
  float* xn96 = (float*)alloc((size_t)BN * 3 * HW96 * 4);
  float* xn48 = (float*)alloc((size_t)BN * 3 * HW48 * 4);
  float* xm96 = (float*)alloc((size_t)BN * 3 * HW96 * 4);
  float* e1q  = (float*)alloc((size_t)BN * ECH * HW96 * 4);
  float* e1s  = (float*)alloc((size_t)BN * ECH * HW48 * 4);
  unsigned short* e1qh = (unsigned short*)alloc((size_t)BN * ECH * HW96 * 2);
  unsigned short* e1sh = (unsigned short*)alloc((size_t)BN * ECH * HW48 * 2);
  float* Eq = (float*)alloc((size_t)BN * ECH * HW96 * 4);
  float* Ek = (float*)alloc((size_t)BN * ECH * HW48 * 4);
  unsigned short* Qb = (unsigned short*)alloc((size_t)BN * NQP * DDIM * 2);
  unsigned short* Kb = (unsigned short*)alloc((size_t)BN * NKP * DDIM * 2);
  float* qn2 = (float*)alloc((size_t)BN * NQP * 4);
  float* kn2 = (float*)alloc((size_t)BN * NKP * 4);
  float* Smat = (float*)alloc((size_t)BN * NQP * NKP * 4);
  int*   idxb = (int*)alloc((size_t)BN * NQ * 5 * 4);
  float* wkb  = (float*)alloc((size_t)BN * NQ * 5 * 4);
  // trunk buffers (96-res)
  float* x0 = (float*)alloc((size_t)BN * NFC * HW96 * 4);
  float* xA = (float*)alloc((size_t)BN * NFC * HW96 * 4);
  float* xB = (float*)alloc((size_t)BN * NFC * HW96 * 4);
  unsigned short* xf16 = (unsigned short*)alloc((size_t)BN * NFC * HW96 * 2);
  unsigned short* tf16 = (unsigned short*)alloc((size_t)BN * NFC * HW96 * 2);
  float* x1ds = (float*)alloc((size_t)BN * NFC * HW48 * 4);
  float* x1lr = (float*)alloc((size_t)BN * NFC * HW96 * 4);
  unsigned short* c96 = (unsigned short*)alloc((size_t)BN * 512 * HW96 * 2);
  // 192-res buffers
  float* G1 = (float*)alloc((size_t)BN * NFC * HW192 * 4);   // x1_hr -> fuse2 out
  float* G2 = (float*)alloc((size_t)BN * 1024 * HW96 * 4);   // up out -> fuse1 out
  float* G3 = (float*)alloc((size_t)BN * NFC * HW192 * 4);   // xu -> f16 temp
  unsigned short* G4 = (unsigned short*)alloc((size_t)BN * 512 * HW192 * 2);

  auto pack = [&](const float* w, unsigned short* dst, int Cin, int Cout) {
    int n = Cin * 9 * Cout;
    k_pack<<<CDIV(n, 256), 256, 0, stream>>>(w, dst, Cin, Cout);
  };
  auto convw = [&](const unsigned short* in, const unsigned short* pk, const float* bias,
                   const float* res, float scale, float* out,
                   int Cin, int Cout, int H, int W, int relu) {
    if (Cin == 64) {
      int cog = Cout / 64;
      dim3 g(cog * (W / 16), H, BN);
      k_conv_wmma<64, 16><<<g, 128, 0, stream>>>(in, pk, bias, res, scale, out,
                                                 Cout, H, W, relu, cog);
    } else if (Cin == 256) {
      int cog = Cout / 128;
      dim3 g(cog * (W / 16), H, BN);
      k_conv_wmma<256, 32><<<g, 128, 0, stream>>>(in, pk, bias, res, scale, out,
                                                  Cout, H, W, relu, cog);
    } else {
      int cog = Cout / 128;
      dim3 g(cog * (W / 16), H, BN);
      k_conv_wmma<512, 32><<<g, 128, 0, stream>>>(in, pk, bias, res, scale, out,
                                                  Cout, H, W, relu, cog);
    }
  };
  auto toh = [&](const float* src, unsigned short* dst, size_t n) {
    k_f32_to_f16<<<CDIV((int)n, 256), 256, 0, stream>>>(src, dst, (int)n, 0);
  };

  // ---- pack all WMMA weights ----
  pack(ew2, pk_e2, ECH, ECH);
  for (int i = 0; i < 8; ++i)
    for (int j = 0; j < 2; ++j)
      pack(b1w + (size_t)(i * 2 + j) * NFC * NFC * 9, pk_b1[i * 2 + j], NFC, NFC);
  pack(fbw1, pk_fb1, 512, NFC);
  pack(fbw2, pk_fb2, NFC, NFC);
  for (int i = 0; i < 8; ++i)
    for (int j = 0; j < 2; ++j)
      pack(b2w + (size_t)(i * 2 + j) * NFC * NFC * 9, pk_b2[i * 2 + j], NFC, NFC);
  pack(b2tw, pk_b2t, NFC, NFC);
  pack(upw, pk_up, NFC, 1024);
  pack(fuw1, pk_fu1, 512, NFC);
  pack(fuw2, pk_fu2, NFC, NFC);

  // ---- embed + graph build ----
  k_preproc<<<CDIV(BN * 3 * (int)HW96, 256), 256, 0, stream>>>(x, xn96, BN * 3 * (int)HW96, (int)HW96, 0);
  k_preproc<<<CDIV(BN * 3 * (int)HW48, 256), 256, 0, stream>>>(x_son, xn48, BN * 3 * (int)HW48, (int)HW48, 0);
  k_conv3_direct<<<CDIV(BN * ECH * (int)HW96, 256), 256, 0, stream>>>(xn96, ew1, eb1, e1q, ECH, 96, 96, 1);
  k_conv3_direct<<<CDIV(BN * ECH * (int)HW48, 256), 256, 0, stream>>>(xn48, ew1, eb1, e1s, ECH, 48, 48, 1);
  toh(e1q, e1qh, (size_t)BN * ECH * HW96);
  toh(e1s, e1sh, (size_t)BN * ECH * HW48);
  convw(e1qh, pk_e2, eb2, nullptr, 0.f, Eq, ECH, ECH, 96, 96, 1);
  convw(e1sh, pk_e2, eb2, nullptr, 0.f, Ek, ECH, ECH, 48, 48, 1);
  k_extract_q<<<CDIV(BN * NQP, 256), 256, 0, stream>>>(Eq, Qb, qn2);
  k_extract_k<<<CDIV(BN * NKP, 256), 256, 0, stream>>>(Ek, Kb, kn2);
  {
    dim3 gq(NKP / 16, NQP / 16, BN);
    k_qk_wmma<<<gq, 32, 0, stream>>>(Qb, Kb, Smat);
  }
  k_topk<<<CDIV(BN * NQ, 256), 256, 0, stream>>>(Smat, qn2, kn2, Qb, Kb, gw, gb, idxb, wkb);

  // ---- head ----
  k_preproc<<<CDIV(BN * 3 * (int)HW96, 256), 256, 0, stream>>>(x, xm96, BN * 3 * (int)HW96, (int)HW96, 1);
  k_conv3_direct<<<CDIV(BN * NFC * (int)HW96, 256), 256, 0, stream>>>(xm96, hdw, hdb, x0, NFC, 96, 96, 0);

  // ---- body1 ----
  const size_t NFE = (size_t)BN * NFC * HW96;
  toh(x0, xf16, NFE);
  const float* cur = x0;
  for (int i = 0; i < 8; ++i) {
    convw(xf16, pk_b1[2 * i], b1b + (size_t)(2 * i) * NFC, nullptr, 0.f, xB, NFC, NFC, 96, 96, 1);
    toh(xB, tf16, NFE);
    convw(tf16, pk_b1[2 * i + 1], b1b + (size_t)(2 * i + 1) * NFC, cur, 0.1f, xA, NFC, NFC, 96, 96, 0);
    cur = xA;
    toh(xA, xf16, NFE);
  }

  // ---- GCN ----
  k_downsample<<<CDIV(BN * NFC * (int)HW48, 256), 256, 0, stream>>>(xA, x1ds);
  k_gcn_lr<<<CDIV(BN * NFC * (int)HW96, 256), 256, 0, stream>>>(x1ds, idxb, wkb, x1lr);
  k_gcn_hr<<<CDIV(BN * NFC * (int)HW192, 256), 256, 0, stream>>>(xA, idxb, wkb, G1);

  // ---- fuse_b ----
  k_concat_f16<<<CDIV(BN * 512 * (int)HW96, 256), 256, 0, stream>>>(xA, x1lr, c96, 96, 96);
  convw(c96, pk_fb1, fbb1, nullptr, 0.f, xB, 512, NFC, 96, 96, 1);
  toh(xB, tf16, NFE);
  convw(tf16, pk_fb2, fbb2, nullptr, 0.f, xA, NFC, NFC, 96, 96, 0);
  toh(xA, xf16, NFE);

  // ---- body2 ----
  for (int i = 0; i < 8; ++i) {
    convw(xf16, pk_b2[2 * i], b2b + (size_t)(2 * i) * NFC, nullptr, 0.f, xB, NFC, NFC, 96, 96, 1);
    toh(xB, tf16, NFE);
    convw(tf16, pk_b2[2 * i + 1], b2b + (size_t)(2 * i + 1) * NFC, xA, 0.1f, xA, NFC, NFC, 96, 96, 0);
    toh(xA, xf16, NFE);
  }
  convw(xf16, pk_b2t, b2tb, x0, 1.0f, x1lr, NFC, NFC, 96, 96, 0);   // x2 in x1lr
  toh(x1lr, xf16, NFE);

  // ---- upsample + fuse_up + tail ----
  convw(xf16, pk_up, upb, nullptr, 0.f, G2, NFC, 1024, 96, 96, 0);
  k_pixel_shuffle<<<CDIV(BN * NFC * (int)HW192, 256), 256, 0, stream>>>(G2, G3);
  k_concat_f16<<<CDIV(BN * 512 * (int)HW192, 256), 256, 0, stream>>>(G3, G1, G4, 192, 192);
  convw(G4, pk_fu1, fub1, nullptr, 0.f, G2, 512, NFC, 192, 192, 1);
  toh(G2, (unsigned short*)G3, (size_t)BN * NFC * HW192);
  convw((unsigned short*)G3, pk_fu2, fub2, nullptr, 0.f, G1, NFC, NFC, 192, 192, 1);
  k_tail<<<CDIV(BN * 3 * (int)HW192, 256), 256, 0, stream>>>(G1, tw, tb, outp);
}
